// RNN_net_36172214567582
// MI455X (gfx1250) — matrix operand order repaired and checked by reading, not compiled, not run on hardware
//
#include <hip/hip_runtime.h>
#include <math.h>

// Problem constants (from the reference)
#define T_STEPS 512
#define BATCH   256
#define DIN     128
#define DH      512
#define DOUT    128

typedef unsigned short u16;
typedef unsigned int   u32;

typedef __attribute__((ext_vector_type(16))) __bf16 v16bf;
typedef __attribute__((ext_vector_type(8)))  __bf16 v8bf;
typedef __attribute__((ext_vector_type(8)))  float  v8f;
typedef int v4i_vec __attribute__((vector_size(16)));   // matches builtin param type

#define AS_GLOBAL __attribute__((address_space(1)))
#define AS_SHARED __attribute__((address_space(3)))

#define KCAT    (DIN + DH)            // 640 concatenated K
#define NKX     (DIN / 32)            // 4 xs k-chunks (h-independent)
#define NKH     (DH / 32)             // 16 h k-chunks
#define NKCH    (KCAT / 32)           // 20 total k-chunks
#define NKCH2   (DH / 32)             // 16 k-chunks (output GEMM)
#define NCT     (DH / 16)             // 32 col tiles (recurrent)
#define NCT2    (DOUT / 16)           // 8 col tiles (output)
#define LDSW    648                   // padded LDS row stride (bf16): 1296 B -> banks spread by 4/row

// Workspace layout (bytes)
#define CNT_OFF    0
#define H0_OFF     256
#define HBYTES     (BATCH * DH * 2)                // 262144
#define H1_OFF     (H0_OFF + HBYTES)
#define WCAT_OFF   (H1_OFF + HBYTES)               // 524544
#define WCAT_BYTES (NKCH * NCT * 32 * 16 * 2)      // 655360
#define W2SW_OFF   (WCAT_OFF + WCAT_BYTES)
#define W2SW_BYTES (NKCH2 * NCT2 * 32 * 16 * 2)    // 131072

#if __has_builtin(__builtin_amdgcn_global_load_async_to_lds_b128)
#define HAVE_ASYNC_LDS 1
#else
#define HAVE_ASYNC_LDS 0
#endif

__device__ __forceinline__ u16 f2bf(float f) {   // RNE f32 -> bf16
  u32 u = __builtin_bit_cast(u32, f);
  u += 0x7FFFu + ((u >> 16) & 1u);
  return (u16)(u >> 16);
}

__device__ __forceinline__ float fast_tanh(float x) {
#if __has_builtin(__builtin_amdgcn_tanhf)
  return __builtin_amdgcn_tanhf(x);        // V_TANH_F32 (gfx1250 transcendental)
#elif __has_builtin(__builtin_amdgcn_tanh_f32)
  return __builtin_amdgcn_tanh_f32(x);
#else
  return tanhf(x);
#endif
}

// ---------------------------------------------------------------------------
// Prep: zero barrier counter + h0; convert W matrices to bf16 in the exact
// per-lane WMMA B-fragment order so the hot loop does contiguous 32B loads.
// B-frag layout (16-bit, K=32): lane<16 holds K=0..7,16..23 of N=lane, lane>=16
// holds K=8..15,24..31 of N=lane-16 (2 bf16 per VGPR, 8 VGPRs).
// ---------------------------------------------------------------------------
__global__ void rnn_prep(const float* __restrict__ W1x,
                         const float* __restrict__ W1h,
                         const float* __restrict__ W2,
                         u16* __restrict__ h0, u16* __restrict__ wcat,
                         u16* __restrict__ w2sw, u32* __restrict__ cnt) {
  const int tid  = blockIdx.x * blockDim.x + threadIdx.x;
  const int nthr = gridDim.x * blockDim.x;
  if (tid == 0) *cnt = 0;
  for (int i = tid; i < BATCH * DH / 2; i += nthr) ((u32*)h0)[i] = 0u;

  // [W1x ; W1h] -> wcat[(kc*32+ct)*512 + lane*16 + i]
  for (int idx = tid; idx < NKCH * NCT * 32 * 16; idx += nthr) {
    int i    = idx & 15;
    int lane = (idx >> 4) & 31;
    int f    = idx >> 9;                 // fragment = kc*32 + ct
    int ct   = f & (NCT - 1);
    int kc   = f >> 5;
    int K    = kc * 32 + ((lane >> 4) << 3) + (i & 7) + ((i >> 3) << 4);
    int N    = ct * 16 + (lane & 15);
    float v  = (K < DIN) ? W1x[K * DH + N] : W1h[(K - DIN) * DH + N];
    wcat[idx] = f2bf(v);
  }
  // W2 -> w2sw[(kc*8+ct)*512 + lane*16 + i]
  for (int idx = tid; idx < NKCH2 * NCT2 * 32 * 16; idx += nthr) {
    int i    = idx & 15;
    int lane = (idx >> 4) & 31;
    int f    = idx >> 9;
    int ct   = f & (NCT2 - 1);
    int kc   = f >> 3;
    int K    = kc * 32 + ((lane >> 4) << 3) + (i & 7) + ((i >> 3) << 4);
    int N    = ct * 16 + (lane & 15);
    w2sw[idx] = f2bf(W2[K * DOUT + N]);
  }
}

// ---------------------------------------------------------------------------
// Persistent recurrent scan: 64 blocks x 8 waves = 512 waves = one 16x16
// output tile per wave per step. Split-phase grid barrier: the h-independent
// xs projection (4 WMMA chunks + f32->bf16 LDS fill) overlaps the wait for
// the previous step's h stores.
// ---------------------------------------------------------------------------
__global__ void __launch_bounds__(256, 1)
rnn_scan(const float* __restrict__ xs, const u16* __restrict__ wcat,
         const float* __restrict__ b1p,
         u16* __restrict__ h0, u16* __restrict__ h1, u32* cnt) {
  __shared__ u16 ldsA[16 * LDSW];                  // [16 rows][640 K] bf16, padded
  const int tid  = threadIdx.x;
  const int lane = tid & 31;
  const int w    = blockIdx.x * 8 + (tid >> 5);
  const int row_tile = w >> 5;                     // same for all waves in block
  const int col_tile = w & 31;
  const int half = lane >> 4;                      // A/B half-wave K split
  const int mrow = lane & 15;
  const int rowbase = row_tile * 16;
  const float b1v = *b1p;
  const u16* bfrag = wcat + (size_t)col_tile * 512 + (size_t)lane * 16;
  const u16* arow  = &ldsA[mrow * LDSW];           // per-lane A row base in LDS

  for (int t = 0; t < T_STEPS; ++t) {
    const u16* hcur = (t & 1) ? h1 : h0;
    u16*       hnxt = (t & 1) ? h0 : h1;
    const float* xs_t = xs + (size_t)t * BATCH * DIN;

    // (1) xs part of A rows -> LDS cols [0,128): h-independent work
    #pragma unroll
    for (int j = 0; j < 2; ++j) {                  // 512 float4 -> bf16x4
      int idx = tid + j * 256;
      int r  = idx >> 5;
      int c4 = idx & 31;
      const float4* src = ((const float4*)(xs_t + (size_t)(rowbase + r) * DIN)) + c4;
      const float4 v = *src;
      __builtin_prefetch((const char*)src + (size_t)BATCH * DIN * 4, 0, 1); // next t
      u32 lo = (u32)f2bf(v.x) | ((u32)f2bf(v.y) << 16);
      u32 hi = (u32)f2bf(v.z) | ((u32)f2bf(v.w) << 16);
      *(uint2*)&ldsA[r * LDSW + c4 * 4] = make_uint2(lo, hi);
    }
    __syncthreads();

    // (2) 4 xs k-chunks of the concatenated GEMM (overlaps barrier wait),
    //     double-buffered A fragments
    v8f acc = {0.f, 0.f, 0.f, 0.f, 0.f, 0.f, 0.f, 0.f};
    {
      v8bf x0[2], x1[2];
      x0[0] = *(const v8bf*)(arow + half * 8);
      x1[0] = *(const v8bf*)(arow + half * 8 + 16);
      #pragma unroll
      for (int kc = 0; kc < NKX; ++kc) {
        const int cur = kc & 1, nxt = cur ^ 1;
        if (kc + 1 < NKX) {
          const int kb = (kc + 1) * 32 + half * 8;
          x0[nxt] = *(const v8bf*)(arow + kb);
          x1[nxt] = *(const v8bf*)(arow + kb + 16);
        }
        v16bf a = __builtin_shufflevector(x0[cur], x1[cur],
                                          0,1,2,3,4,5,6,7,8,9,10,11,12,13,14,15);
        v16bf b = *(const v16bf*)(bfrag + (size_t)kc * 32 * 512);
        acc = __builtin_amdgcn_wmma_f32_16x16x32_bf16(false, a, false, b,
                                                      (short)0, acc, false, false);
      }
    }

    // (3) split-barrier WAIT: previous step's h stores visible device-wide
    if (t > 0) {
      if (tid == 0) {
        const u32 target = (u32)t * gridDim.x;
        while (__hip_atomic_load(cnt, __ATOMIC_RELAXED,
                                 __HIP_MEMORY_SCOPE_AGENT) < target)
          __builtin_amdgcn_s_sleep(1);
      }
      __syncthreads();
      __threadfence();                             // acquire: drop stale h lines
    }

    // (4) h part of A rows -> LDS cols [128,640)
#if HAVE_ASYNC_LDS
    #pragma unroll
    for (int j = 0; j < 4; ++j) {                  // 1024 x 16B async copies
      int idx = tid + j * 256;
      int r  = idx >> 6;
      int c8 = idx & 63;
      const u16* g = hcur + (size_t)(rowbase + r) * DH + c8 * 8;
      u16* l = &ldsA[r * LDSW + DIN + c8 * 8];
      __builtin_amdgcn_global_load_async_to_lds_b128(
          (AS_GLOBAL v4i_vec*)g, (AS_SHARED v4i_vec*)l, 0, 0);
    }
    asm volatile("s_wait_asynccnt 0x0" ::: "memory");
#else
    #pragma unroll
    for (int j = 0; j < 4; ++j) {                  // 1024 x 16B copies
      int idx = tid + j * 256;
      int r  = idx >> 6;
      int c8 = idx & 63;
      const uint4 v = *(const uint4*)(hcur + (size_t)(rowbase + r) * DH + c8 * 8);
      *(uint4*)&ldsA[r * LDSW + DIN + c8 * 8] = v;
    }
#endif
    __syncthreads();

    // (5) 16 h k-chunks, double-buffered A fragments so ds_load(k+1)
    //     overlaps wmma(k) instead of s_wait_dscnt 0 serialization
    v8bf a0[2], a1[2];
    a0[0] = *(const v8bf*)(arow + DIN + half * 8);
    a1[0] = *(const v8bf*)(arow + DIN + half * 8 + 16);
    #pragma unroll
    for (int kc = 0; kc < NKH; ++kc) {
      const int cur = kc & 1, nxt = cur ^ 1;
      if (kc + 1 < NKH) {
        const int kb = DIN + (kc + 1) * 32 + half * 8;
        a0[nxt] = *(const v8bf*)(arow + kb);
        a1[nxt] = *(const v8bf*)(arow + kb + 16);
      }
      v16bf a = __builtin_shufflevector(a0[cur], a1[cur],
                                        0,1,2,3,4,5,6,7,8,9,10,11,12,13,14,15);
      v16bf b = *(const v16bf*)(bfrag + (size_t)(NKX + kc) * 32 * 512);
      acc = __builtin_amdgcn_wmma_f32_16x16x32_bf16(false, a, false, b,
                                                    (short)0, acc, false, false);
    }

    // (6) h_new = tanh(acc + b1) -> bf16 NT scatter (consumed by other WGPs
    //     out of L2; C layout: n=mrow, m=v+8*half)
    #pragma unroll
    for (int v = 0; v < 8; ++v) {
      int m = v + half * 8;
      float x = acc[v] + b1v;
      __builtin_nontemporal_store(
          f2bf(fast_tanh(x)),
          &hnxt[(size_t)(rowbase + m) * DH + col_tile * 16 + mrow]);
    }

    // (7) split-barrier SIGNAL (no wait here; next step overlaps it)
    __threadfence();                               // release h stores
    __syncthreads();
    if (tid == 0)
      __hip_atomic_fetch_add(cnt, 1u, __ATOMIC_RELAXED, __HIP_MEMORY_SCOPE_AGENT);
  }
}

// ---------------------------------------------------------------------------
// Output GEMM: out = h_final @ W2 + b2, [256x512]x[512x128] -> 128 tiles.
// 16 blocks x 8 waves = 128 waves, one tile each.
// ---------------------------------------------------------------------------
__global__ void __launch_bounds__(256)
rnn_out(const u16* __restrict__ hfin, const u16* __restrict__ w2sw,
        const float* __restrict__ b2p, float* __restrict__ out) {
  const int tid  = threadIdx.x;
  const int lane = tid & 31;
  const int w    = blockIdx.x * 8 + (tid >> 5);
  const int row_tile = w >> 3;
  const int col_tile = w & 7;
  const int half = lane >> 4;
  const int mrow = lane & 15;
  const float b2v = *b2p;

  v8f acc = {0.f, 0.f, 0.f, 0.f, 0.f, 0.f, 0.f, 0.f};
  #pragma unroll
  for (int kc = 0; kc < NKCH2; ++kc) {
    const int kb = kc * 32 + half * 8;
    const u16* ap = hfin + (size_t)(row_tile * 16 + mrow) * DH + kb;
    v8bf a0 = *(const v8bf*)ap;
    v8bf a1 = *(const v8bf*)(ap + 16);
    v16bf a = __builtin_shufflevector(a0, a1, 0,1,2,3,4,5,6,7,8,9,10,11,12,13,14,15);
    v16bf b = *(const v16bf*)(w2sw + ((size_t)(kc * NCT2 + col_tile) * 32 + lane) * 16);
    acc = __builtin_amdgcn_wmma_f32_16x16x32_bf16(false, a, false, b,
                                                  (short)0, acc, false, false);
  }
  #pragma unroll
  for (int v = 0; v < 8; ++v) {
    int m = v + half * 8;
    out[(size_t)(row_tile * 16 + m) * DOUT + col_tile * 16 + mrow] = acc[v] + b2v;
  }
}

extern "C" void kernel_launch(void* const* d_in, const int* in_sizes, int n_in,
                              void* d_out, int out_size, void* d_ws, size_t ws_size,
                              hipStream_t stream) {
  (void)in_sizes; (void)n_in; (void)out_size; (void)ws_size;
  const float* xs  = (const float*)d_in[0];
  const float* W1x = (const float*)d_in[1];
  const float* W1h = (const float*)d_in[2];
  const float* b1  = (const float*)d_in[3];
  const float* W2  = (const float*)d_in[4];
  const float* b2  = (const float*)d_in[5];

  char* ws  = (char*)d_ws;
  u32*  cnt = (u32*)(ws + CNT_OFF);
  u16*  h0  = (u16*)(ws + H0_OFF);
  u16*  h1  = (u16*)(ws + H1_OFF);
  u16*  wct = (u16*)(ws + WCAT_OFF);
  u16*  w2s = (u16*)(ws + W2SW_OFF);

  rnn_prep<<<512, 256, 0, stream>>>(W1x, W1h, W2, h0, wct, w2s, cnt);
  rnn_scan<<<64, 256, 0, stream>>>(xs, wct, b1, h0, h1, cnt);
  // T_STEPS is even -> final h lives in buffer h0
  rnn_out<<<16, 256, 0, stream>>>(h0, w2s, b2, (float*)d_out);
}
